// STGAT_42700564857551
// MI455X (gfx1250) — matrix operand order repaired and checked by reading, compiled, not run on hardware
//
#include <hip/hip_runtime.h>
#include <hip/hip_bf16.h>
#include <math.h>

// ST-GAT for MI455X (gfx1250, wave32). Dense projections via v_wmma_f32_16x16x32_f16
// (fp32 inputs converted to f16 fragments, f32 accumulate). Edge passes rely on the
// 192MB L2 holding the 38.4MB node-feature tables, with f32 atomics for segment ops.

typedef __attribute__((ext_vector_type(16))) _Float16 v16h;
typedef __attribute__((ext_vector_type(8)))  float    v8f;

#define HID     64
#define TFRAMES 100

// ---------- helpers ----------
__device__ __forceinline__ unsigned enc_mono(float f) {
    unsigned u = __float_as_uint(f);
    return (u >> 31) ? ~u : (u | 0x80000000u);   // order-preserving float->uint
}
__device__ __forceinline__ float dec_mono(unsigned e) {
    return (e & 0x80000000u) ? __uint_as_float(e ^ 0x80000000u) : __uint_as_float(~e);
}
__device__ __forceinline__ v16h load16h(const float* __restrict__ p) {
    float4 f0 = *(const float4*)(p + 0);
    float4 f1 = *(const float4*)(p + 4);
    float4 f2 = *(const float4*)(p + 8);
    float4 f3 = *(const float4*)(p + 12);
    v16h v;
    v[0]=(_Float16)f0.x;  v[1]=(_Float16)f0.y;  v[2]=(_Float16)f0.z;  v[3]=(_Float16)f0.w;
    v[4]=(_Float16)f1.x;  v[5]=(_Float16)f1.y;  v[6]=(_Float16)f1.z;  v[7]=(_Float16)f1.w;
    v[8]=(_Float16)f2.x;  v[9]=(_Float16)f2.y;  v[10]=(_Float16)f2.z; v[11]=(_Float16)f2.w;
    v[12]=(_Float16)f3.x; v[13]=(_Float16)f3.y; v[14]=(_Float16)f3.z; v[15]=(_Float16)f3.w;
    return v;
}

__global__ void k_zero(float* __restrict__ p, size_t n) {
    size_t i = (size_t)blockIdx.x * blockDim.x + threadIdx.x;
    if (i < n) p[i] = 0.f;
}

// ---------- self-loop edge attr (PyG fill_value='mean') ----------
__global__ void k_loop_accum(const int* __restrict__ edst, const float* __restrict__ eattr,
                             float* __restrict__ cnt, float* __restrict__ lsum, int E) {
    int e = blockIdx.x * blockDim.x + threadIdx.x;
    if (e >= E) return;
    int d = edst[e];
    atomicAdd(&cnt[d], 1.f);
    #pragma unroll
    for (int k = 0; k < 16; k++)
        atomicAdd(&lsum[(size_t)d * 16 + k], eattr[(size_t)e * 16 + k]);
}
__global__ void k_loop_div(float* __restrict__ lsum, const float* __restrict__ cnt, int Nn) {
    int i = blockIdx.x * blockDim.x + threadIdx.x;
    if (i >= Nn * 16) return;
    lsum[i] /= fmaxf(cnt[i >> 4], 1.f);
}

// ---------- dual-output WMMA GEMM: O1 = A@W1, O2 = A@W2 ----------
// One wave per 16x16 output tile; A fragment reused for both WMMAs.
// A: rows x K (fp32, row-major), W: K x Ncols (fp32, row-major). K%32==0, Ncols%16==0.
__global__ void __launch_bounds__(256)
k_gemm_dual(const float* __restrict__ A,
            const float* __restrict__ W1, const float* __restrict__ W2,
            float* __restrict__ O1, float* __restrict__ O2,
            int tilesM, int tilesN, int K, int Ncols) {
    int wave = (int)((blockIdx.x * blockDim.x + threadIdx.x) >> 5);
    int lane = threadIdx.x & 31;
    if (wave >= tilesM * tilesN) return;
    int tm = wave / tilesN, tn = wave % tilesN;

    v8f cl = {}; v8f cr = {};
    int rm = lane & 15;              // row within tile
    int kb = (lane >> 4) * 8;        // per-lane K sub-base (ISA A-fragment layout)
    const float* arow = A + (size_t)(tm * 16 + rm) * K + kb;

    for (int kt = 0; kt < K; kt += 32) {
        // A fragment 16x32 f16: lane<16 holds K={kb..kb+7, kb+16..kb+23}
        const float* ab = arow + kt;
        float4 f0 = *(const float4*)(ab + 0);
        float4 f1 = *(const float4*)(ab + 4);
        float4 f2 = *(const float4*)(ab + 16);
        float4 f3 = *(const float4*)(ab + 20);
        v16h a;
        a[0]=(_Float16)f0.x;  a[1]=(_Float16)f0.y;  a[2]=(_Float16)f0.z;  a[3]=(_Float16)f0.w;
        a[4]=(_Float16)f1.x;  a[5]=(_Float16)f1.y;  a[6]=(_Float16)f1.z;  a[7]=(_Float16)f1.w;
        a[8]=(_Float16)f2.x;  a[9]=(_Float16)f2.y;  a[10]=(_Float16)f2.z; a[11]=(_Float16)f2.w;
        a[12]=(_Float16)f3.x; a[13]=(_Float16)f3.y; a[14]=(_Float16)f3.z; a[15]=(_Float16)f3.w;

        // B fragment 32x16 f16: lane = column N (lane&15), elements = K in same order as A
        v16h bl, br;
        #pragma unroll
        for (int s = 0; s < 16; s++) {
            size_t widx = (size_t)(kt + kb + (s < 8 ? s : s + 8)) * Ncols + tn * 16 + rm;
            bl[s] = (_Float16)W1[widx];
            br[s] = (_Float16)W2[widx];
        }

        cl = __builtin_amdgcn_wmma_f32_16x16x32_f16(false, a, false, bl, (short)0, cl, false, false);
        cr = __builtin_amdgcn_wmma_f32_16x16x32_f16(false, a, false, br, (short)0, cr, false, false);
    }
    // D layout: VGPR r -> M = r (+8 for lanes 16..31), N = lane&15
    int row0 = tm * 16 + ((lane >> 4) ? 8 : 0);
    int col  = tn * 16 + rm;
    #pragma unroll
    for (int r = 0; r < 8; r++) {
        O1[(size_t)(row0 + r) * Ncols + col] = cl[r];
        O2[(size_t)(row0 + r) * Ncols + col] = cr[r];
    }
}

// ---------- edge pass A: attention scores + segment max ----------
// One wave per edge; edge-transform e=ea@We fused via LDS-cached We.
__global__ void __launch_bounds__(256)
k_edge_scores(const float* __restrict__ xl, const float* __restrict__ xr,
              const float* __restrict__ We, const float* __restrict__ att,
              const int* __restrict__ esrc, const int* __restrict__ edst,
              const float* __restrict__ eattr, const float* __restrict__ lattr,
              float* __restrict__ scores, unsigned* __restrict__ smax,
              int E, int Nn, int H) {
    const int HC = H * 64;
    __shared__ float sWe[16 * 192];
    __shared__ float sAtt[192];
    for (int t = threadIdx.x; t < 16 * HC; t += blockDim.x) sWe[t] = We[t];
    for (int t = threadIdx.x; t < HC; t += blockDim.x) sAtt[t] = att[t];
    __syncthreads();

    int lane = threadIdx.x & 31;
    int e = blockIdx.x * (blockDim.x >> 5) + (threadIdx.x >> 5);
    int E2 = E + Nn;
    if (e >= E2) return;
    int s, d; const float* ea;
    if (e < E) { s = esrc[e]; d = edst[e]; ea = eattr + (size_t)e * 16; }
    else       { s = d = e - E;            ea = lattr + (size_t)(e - E) * 16; }

    float eav[16];
    #pragma unroll
    for (int k = 0; k < 16; k++) eav[k] = ea[k];

    float sc[3] = {0.f, 0.f, 0.f};
    const float* xls = xl + (size_t)s * HC;
    const float* xrd = xr + (size_t)d * HC;
    for (int idx = lane; idx < HC; idx += 32) {
        float et = 0.f;
        #pragma unroll
        for (int k = 0; k < 16; k++) et += eav[k] * sWe[k * HC + idx];
        float m = xls[idx] + xrd[idx] + et;
        float lr = m > 0.f ? m : 0.2f * m;          // leaky_relu(0.2)
        sc[idx >> 6] += sAtt[idx] * lr;
    }
    #pragma unroll
    for (int h = 0; h < 3; h++)
        #pragma unroll
        for (int off = 16; off; off >>= 1)
            sc[h] += __shfl_xor(sc[h], off, 32);
    if (lane < H) {
        scores[(size_t)e * H + lane] = sc[lane];
        atomicMax(&smax[(size_t)d * H + lane], enc_mono(sc[lane]));
    }
}

// ---------- edge pass B: exp + segment sum ----------
__global__ void k_edge_norm(const int* __restrict__ edst, float* __restrict__ scores,
                            const unsigned* __restrict__ smax, float* __restrict__ denom,
                            int E, int Nn, int H) {
    int e = blockIdx.x * blockDim.x + threadIdx.x;
    int E2 = E + Nn;
    if (e >= E2) return;
    int d = (e < E) ? edst[e] : (e - E);
    for (int h = 0; h < H; h++) {
        float mx = dec_mono(smax[(size_t)d * H + h]);
        float ex = expf(scores[(size_t)e * H + h] - mx);
        scores[(size_t)e * H + h] = ex;
        atomicAdd(&denom[(size_t)d * H + h], ex);
    }
}

// ---------- edge pass C: alpha-weighted scatter-add of xl[src] ----------
__global__ void __launch_bounds__(256)
k_edge_agg(const float* __restrict__ xl, const int* __restrict__ esrc,
           const int* __restrict__ edst, const float* __restrict__ scores,
           const float* __restrict__ denom, float* __restrict__ agg,
           int E, int Nn, int H) {
    int lane = threadIdx.x & 31;
    int e = blockIdx.x * (blockDim.x >> 5) + (threadIdx.x >> 5);
    int E2 = E + Nn; const int HC = H * 64;
    if (e >= E2) return;
    int s, d;
    if (e < E) { s = esrc[e]; d = edst[e]; } else { s = d = e - E; }
    float alpha[3];
    for (int h = 0; h < H; h++)
        alpha[h] = scores[(size_t)e * H + h] / denom[(size_t)d * H + h];
    const float* xls = xl + (size_t)s * HC;
    float* ad = agg + (size_t)d * HC;
    for (int idx = lane; idx < HC; idx += 32)
        atomicAdd(&ad[idx], alpha[idx >> 6] * xls[idx]);
}

// ---------- bias + ELU (in place) ----------
__global__ void k_bias_elu(float* __restrict__ a, const float* __restrict__ bias, int n, int HC) {
    int i = blockIdx.x * blockDim.x + threadIdx.x;
    if (i >= n * HC) return;
    float v = a[i] + bias[i % HC];
    a[i] = v > 0.f ? v : expm1f(v);
}

// ---------- global mean pool (sum + count) ----------
__global__ void k_pool(const float* __restrict__ h2, const int* __restrict__ batch,
                       float* __restrict__ feats, float* __restrict__ nper, int Nn) {
    int i = blockIdx.x * blockDim.x + threadIdx.x;
    if (i >= Nn) return;
    int b = batch[i];
    atomicAdd(&nper[b], 1.f);
    for (int c = 0; c < HID; c++)
        atomicAdd(&feats[b * HID + c], h2[(size_t)i * HID + c]);
}

// ---------- LSTM over frames + MLP head (single block, 256 = 4 gates x 64 units) ----------
__global__ void __launch_bounds__(256)
k_lstm_head(const float* __restrict__ feats, const float* __restrict__ nper,
            const float* __restrict__ Wih, const float* __restrict__ Whh,
            const float* __restrict__ bih, const float* __restrict__ bhh,
            const float* __restrict__ Wlin1, const float* __restrict__ blin1,
            const float* __restrict__ Wlin2, const float* __restrict__ blin2,
            float* __restrict__ out) {
    __shared__ float sx[HID], sh[HID], sz[4 * HID];
    int tid = threadIdx.x;
    float creg = 0.f;
    if (tid < HID) sh[tid] = 0.f;
    __syncthreads();
    for (int t = 0; t < TFRAMES; t++) {
        if (tid < HID) sx[tid] = feats[t * HID + tid] / nper[t];
        __syncthreads();
        float acc = bih[tid] + bhh[tid];
        const float* wi = Wih + tid * HID;
        const float* wh = Whh + tid * HID;
        for (int k = 0; k < HID; k++) acc += wi[k] * sx[k] + wh[k] * sh[k];
        sz[tid] = acc;
        __syncthreads();
        if (tid < HID) {
            float ig = 1.f / (1.f + expf(-sz[tid]));
            float fg = 1.f / (1.f + expf(-sz[HID + tid]));
            float gg = tanhf(sz[2 * HID + tid]);
            float og = 1.f / (1.f + expf(-sz[3 * HID + tid]));
            creg = fg * creg + ig * gg;
            sh[tid] = og * tanhf(creg);
        }
        __syncthreads();
    }
    if (tid < 32) {
        float acc = blin1[tid];
        for (int k = 0; k < HID; k++) acc += sh[k] * Wlin1[k * 32 + tid];
        sz[tid] = fmaxf(acc, 0.f) * Wlin2[tid];
    }
    __syncthreads();
    if (tid == 0) {
        float s = blin2[0];
        for (int j = 0; j < 32; j++) s += sz[j];
        out[0] = s;
    }
}

extern "C" void kernel_launch(void* const* d_in, const int* in_sizes, int n_in,
                              void* d_out, int out_size, void* d_ws, size_t ws_size,
                              hipStream_t stream) {
    (void)n_in; (void)out_size; (void)ws_size;
    const float* x     = (const float*)d_in[0];
    const int*   esrc  = (const int*)d_in[1];
    const int*   edst  = (const int*)d_in[2];
    const float* eattr = (const float*)d_in[3];
    const int*   batch = (const int*)d_in[4];
    const float* Wl1 = (const float*)d_in[5];
    const float* Wr1 = (const float*)d_in[6];
    const float* We1 = (const float*)d_in[7];
    const float* att1= (const float*)d_in[8];
    const float* b1  = (const float*)d_in[9];
    const float* Wl2 = (const float*)d_in[10];
    const float* Wr2 = (const float*)d_in[11];
    const float* We2 = (const float*)d_in[12];
    const float* att2= (const float*)d_in[13];
    const float* b2  = (const float*)d_in[14];
    const float* Wih = (const float*)d_in[15];
    const float* Whh = (const float*)d_in[16];
    const float* bih = (const float*)d_in[17];
    const float* bhh = (const float*)d_in[18];
    const float* Wlin1=(const float*)d_in[19];
    const float* blin1=(const float*)d_in[20];
    const float* Wlin2=(const float*)d_in[21];
    const float* blin2=(const float*)d_in[22];

    const int N  = in_sizes[4];     // nodes
    const int E  = in_sizes[1];     // edges
    const int E2 = E + N;           // with self loops

    // workspace layout (256B-aligned regions)
    float* ws = (float*)d_ws;
    size_t off = 0;
    auto alloc = [&](size_t n) { float* p = ws + off; off += (n + 63) & ~(size_t)63; return p; };
    float*    cnt    = alloc(N);
    float*    lattr  = alloc((size_t)N * 16);
    float*    xl     = alloc((size_t)N * 192);   // layer1 xl; reused as layer2 xl (N*64)
    float*    xr     = alloc((size_t)N * 192);   // layer1 xr; reused as layer2 xr
    float*    agg    = alloc((size_t)N * 192);   // layer1 agg -> h1; reused as layer2 agg -> h2
    float*    scores = alloc((size_t)E2 * 3);
    unsigned* smax   = (unsigned*)alloc((size_t)N * 3);
    float*    denom  = alloc((size_t)N * 3);
    float*    feats  = alloc((size_t)TFRAMES * HID);
    float*    nper   = alloc(TFRAMES);

    auto zero = [&](float* p, size_t n) {
        k_zero<<<(unsigned)((n + 255) / 256), 256, 0, stream>>>(p, n);
    };
    const int tilesM = N / 16;
    const int ewaveBlocks = (E2 + 7) / 8;

    // self-loop attrs
    zero(cnt, N); zero(lattr, (size_t)N * 16);
    k_loop_accum<<<(E + 255) / 256, 256, 0, stream>>>(edst, eattr, cnt, lattr, E);
    k_loop_div<<<(N * 16 + 255) / 256, 256, 0, stream>>>(lattr, cnt, N);

    // ---- layer 1 (heads=3, HC=192) ----
    k_gemm_dual<<<(tilesM * 12 + 7) / 8, 256, 0, stream>>>(x, Wl1, Wr1, xl, xr, tilesM, 12, 128, 192);
    zero((float*)smax, (size_t)N * 3); zero(denom, (size_t)N * 3); zero(agg, (size_t)N * 192);
    k_edge_scores<<<ewaveBlocks, 256, 0, stream>>>(xl, xr, We1, att1, esrc, edst, eattr, lattr,
                                                   scores, smax, E, N, 3);
    k_edge_norm<<<(E2 + 255) / 256, 256, 0, stream>>>(edst, scores, smax, denom, E, N, 3);
    k_edge_agg<<<ewaveBlocks, 256, 0, stream>>>(xl, esrc, edst, scores, denom, agg, E, N, 3);
    k_bias_elu<<<(N * 192 + 255) / 256, 256, 0, stream>>>(agg, b1, N, 192);   // agg = h1

    // ---- layer 2 (heads=1, HC=64) ----
    k_gemm_dual<<<(tilesM * 4 + 7) / 8, 256, 0, stream>>>(agg, Wl2, Wr2, xl, xr, tilesM, 4, 192, 64);
    zero((float*)smax, N); zero(denom, N); zero(agg, (size_t)N * 64);
    k_edge_scores<<<ewaveBlocks, 256, 0, stream>>>(xl, xr, We2, att2, esrc, edst, eattr, lattr,
                                                   scores, smax, E, N, 1);
    k_edge_norm<<<(E2 + 255) / 256, 256, 0, stream>>>(edst, scores, smax, denom, E, N, 1);
    k_edge_agg<<<ewaveBlocks, 256, 0, stream>>>(xl, esrc, edst, scores, denom, agg, E, N, 1);
    k_bias_elu<<<(N * 64 + 255) / 256, 256, 0, stream>>>(agg, b2, N, 64);     // agg = h2

    // ---- pool + LSTM + head ----
    zero(feats, (size_t)TFRAMES * HID); zero(nper, TFRAMES);
    k_pool<<<(N + 255) / 256, 256, 0, stream>>>(agg, batch, feats, nper, N);
    k_lstm_head<<<1, 256, 0, stream>>>(feats, nper, Wih, Whh, bih, bhh,
                                       Wlin1, blin1, Wlin2, blin2, (float*)d_out);
}
